// Qwen2MoEMLPLayer_9655086482009
// MI455X (gfx1250) — matrix operand order repaired and checked by reading, not compiled, not run on hardware
//
#include <hip/hip_runtime.h>
#include <hip/hip_bf16.h>

// ---------------- problem constants (match reference) ----------------
constexpr int S_   = 2048;
constexpr int B_   = 4;
constexpr int HID  = 1024;          // hidden size
constexpr int T_   = S_ * B_;       // 8192 tokens
constexpr int NEXP = 64;            // experts
constexpr int TOPK = 4;
constexpr int IMID = 512;           // moe intermediate
constexpr int N1   = 2 * IMID;      // fc1 output width (gate|up) = 1024
constexpr int CAPE = 1024;          // per-expert capacity

constexpr int AS = 40;              // LDS A row stride (shorts): 64B data + 16B pad
constexpr int BS = 40;              // LDS B row stride (shorts)

// ---------------- workspace layout (bytes) ----------------
constexpr size_t MB = 1024ull * 1024ull;
constexpr size_t OFF_CNT   = 0;                                   // 64 ints
constexpr size_t OFF_SMAP  = 1 * MB;                              // T*K ints
constexpr size_t OFF_SPROB = 2 * MB;                              // T*K floats
constexpr size_t OFF_W1B   = 4 * MB;                              // E*H*2I bf16 (128MB)
constexpr size_t OFF_W2B   = OFF_W1B + (size_t)NEXP * HID * N1 * 2;     // +64MB
constexpr size_t OFF_BUF   = OFF_W2B + (size_t)NEXP * IMID * HID * 2;   // E*CAP*H bf16
constexpr size_t OFF_G     = OFF_BUF + (size_t)NEXP * CAPE * HID * 2;   // E*CAP*I bf16
constexpr size_t OFF_Y     = OFF_G   + (size_t)NEXP * CAPE * IMID * 2;  // E*CAP*H bf16

// ---------------- small helpers ----------------
__device__ __forceinline__ unsigned short f2bf(float f) {
    union { float f; unsigned u; } v; v.f = f;
    unsigned r = v.u + 0x7fffu + ((v.u >> 16) & 1u);   // round-to-nearest-even
    return (unsigned short)(r >> 16);
}
__device__ __forceinline__ float bf2f(unsigned short u) {
    union { unsigned u; float f; } v; v.u = ((unsigned)u) << 16;
    return v.f;
}

typedef __attribute__((ext_vector_type(16))) __bf16 bf16x16;
typedef __attribute__((ext_vector_type(8)))  float  f32x8;
typedef __attribute__((ext_vector_type(4)))  unsigned int u32x4;
typedef __attribute__((ext_vector_type(8)))  int i32x8;
typedef __attribute__((ext_vector_type(4)))  int i32x4;
union FragBF { bf16x16 v; unsigned short u[16]; };

// ---- CDNA5 async global->LDS copy (16B per lane), tracked by ASYNCcnt ----
__device__ __forceinline__ void async_ld16(void* lds_ptr, const void* gptr) {
    // generic LDS pointer: low 32 bits == LDS byte offset
    unsigned loff = (unsigned)(size_t)lds_ptr;
    asm volatile("global_load_async_to_lds_b128 %0, %1, off"
                 :: "v"(loff), "v"(gptr) : "memory");
}
__device__ __forceinline__ void wait_async() {
    asm volatile("s_wait_asynccnt 0x0" ::: "memory");
}

// ---- CDNA5 Tensor Data Mover: 2D tile (tileW x tileH, 2B elems) -> LDS ----
// Hardware pad: every 16 DWORDs (one 64B row, tileW=32) insert 4 DWORDs (16B)
// => LDS row stride of 80B == AS shorts, matching the fragment loaders.
__device__ __forceinline__ void tdm_load_2d(unsigned lds_off, const void* gptr,
                                            unsigned rowStrideElems,
                                            unsigned tdim0, unsigned tdim1,
                                            unsigned tileW, unsigned tileH) {
    unsigned long long ga = (unsigned long long)gptr;
    u32x4 g0;
    g0[0] = 1u;                                            // count=1, user mode
    g0[1] = lds_off;                                       // lds_addr [63:32]
    g0[2] = (unsigned)ga;                                  // global_addr lo
    g0[3] = ((unsigned)(ga >> 32) & 0x01FFFFFFu)           // global_addr [56:32]
            | 0x80000000u;                                 // type=2 ("image")
    i32x8 g1;
    g1[0] = (int)((1u << 16)                               // data_size = 2B
                | (1u << 20)                               // pad_enable
                | (3u << 22)                               // pad_interval: 16 DW
                | (3u << 25));                             // pad_amount: 4 DW
    g1[1] = (int)((tdim0 & 0xFFFFu) << 16);                // tensor_dim0 [15:0]
    g1[2] = (int)((tdim0 >> 16) | ((tdim1 & 0xFFFFu) << 16)); // dim0 hi | dim1 lo
    g1[3] = (int)((tdim1 >> 16) | (tileW << 16));          // dim1 hi | tile_dim0
    g1[4] = (int)tileH;                                    // tile_dim1 (tile_dim2=0)
    g1[5] = (int)rowStrideElems;                           // tensor_dim0_stride lo
    g1[6] = 0;                                             // stride hi | dim1_stride lo
    g1[7] = 0;
    i32x4 z4 = {0, 0, 0, 0};                               // groups 2/3: unused (2D)
    i32x8 z8 = {0, 0, 0, 0, 0, 0, 0, 0};
    __builtin_amdgcn_tensor_load_to_lds(g0, g1, z4, z4, z8, 0);
}
__device__ __forceinline__ void wait_tensor() {
    __builtin_amdgcn_s_wait_tensorcnt((short)0);
}

// A-matrix 16x32 bf16 fragment from row-major LDS tile [rows][AS]
// lanes 0-15: V0..3 K=0..7, V4..7 K=16..23 ; lanes 16-31: +8
__device__ __forceinline__ bf16x16 load_fragA(const unsigned short* lds, int row, int lane) {
    FragBF f;
    const int kb = (lane < 16) ? 0 : 8;
    const unsigned short* p = lds + row * AS;
#pragma unroll
    for (int v = 0; v < 8; ++v) {
        int k0 = (v < 4) ? (kb + 2 * v) : (16 + kb + 2 * (v - 4));
        f.u[2 * v]     = p[k0];
        f.u[2 * v + 1] = p[k0 + 1];
    }
    return f.v;
}
// B-matrix 32x16 bf16 fragment from K-transposed LDS tile Bs[n][BS]
// lanes 0-15: K=0..7,16..23 ; lanes 16-31: +8 ; N = lane&15
__device__ __forceinline__ bf16x16 load_fragB(const unsigned short* lds, int col, int lane) {
    FragBF f;
    const int kb = (lane < 16) ? 0 : 8;
    const unsigned short* p = lds + col * BS;
#pragma unroll
    for (int v = 0; v < 8; ++v) {
        int k0 = (v < 4) ? (kb + 2 * v) : (16 + kb + 2 * (v - 4));
        f.u[2 * v]     = p[k0];
        f.u[2 * v + 1] = p[k0 + 1];
    }
    return f.v;
}
__device__ __forceinline__ f32x8 wmma_bf16(bf16x16 a, bf16x16 b, f32x8 c) {
    return __builtin_amdgcn_wmma_f32_16x16x32_bf16(false, a, false, b, (short)0, c,
                                                   false, false);
}

// stage a 128x32 bf16 A slab into LDS via per-lane async copies (whole workgroup)
__device__ __forceinline__ void stageA_async(unsigned short* Asel,
                                             const unsigned short* src, size_t rowStride,
                                             int tid, int kk) {
    const int r = tid >> 1, half = tid & 1;
    const unsigned short* gp = src + (size_t)r * rowStride + kk + half * 16;
    unsigned short* lp = Asel + r * AS + half * 16;
    async_ld16(lp, gp);
    async_ld16(lp + 8, gp + 8);
}

// ---------------- kernel 1: weight fp32->bf16 conversion + counter init ----------------
__global__ __launch_bounds__(256) void convert_init_kernel(
    const float* __restrict__ w1, const float* __restrict__ w2,
    unsigned short* __restrict__ w1b, unsigned short* __restrict__ w2b,
    int* __restrict__ cnt) {
    const size_t gid    = (size_t)blockIdx.x * blockDim.x + threadIdx.x;
    const size_t stride = (size_t)gridDim.x * blockDim.x;
    if (gid < NEXP) cnt[gid] = 0;

    const size_t n1 = (size_t)NEXP * HID * N1 / 4;
    for (size_t i = gid; i < n1; i += stride) {
        float4 v = ((const float4*)w1)[i];
        ushort4 o;
        o.x = f2bf(v.x); o.y = f2bf(v.y); o.z = f2bf(v.z); o.w = f2bf(v.w);
        ((ushort4*)w1b)[i] = o;
    }
    const size_t n2 = (size_t)NEXP * IMID * HID / 4;
    for (size_t i = gid; i < n2; i += stride) {
        float4 v = ((const float4*)w2)[i];
        ushort4 o;
        o.x = f2bf(v.x); o.y = f2bf(v.y); o.z = f2bf(v.z); o.w = f2bf(v.w);
        ((ushort4*)w2b)[i] = o;
    }
}

// ---------------- kernel 2: router (softmax + top-4) + token scatter ----------------
__global__ __launch_bounds__(256) void router_kernel(
    const float* __restrict__ x, const float* __restrict__ rw,
    int* __restrict__ cnt, int* __restrict__ smap, float* __restrict__ sprob,
    unsigned short* __restrict__ buf) {
    const int lane = threadIdx.x & 31;
    const int wv   = threadIdx.x >> 5;
    const int t    = blockIdx.x * 8 + wv;         // one wave per token
    const float* xt = x + (size_t)t * HID;

    const int e0 = lane * 2, e1 = lane * 2 + 1;   // 2 experts per lane
    float acc0 = 0.f, acc1 = 0.f;
    for (int h = 0; h < HID; h += 4) {
        float4 xv = *(const float4*)(xt + h);
        acc0 = fmaf(xv.x, rw[(h + 0) * NEXP + e0], acc0);
        acc1 = fmaf(xv.x, rw[(h + 0) * NEXP + e1], acc1);
        acc0 = fmaf(xv.y, rw[(h + 1) * NEXP + e0], acc0);
        acc1 = fmaf(xv.y, rw[(h + 1) * NEXP + e1], acc1);
        acc0 = fmaf(xv.z, rw[(h + 2) * NEXP + e0], acc0);
        acc1 = fmaf(xv.z, rw[(h + 2) * NEXP + e1], acc1);
        acc0 = fmaf(xv.w, rw[(h + 3) * NEXP + e0], acc0);
        acc1 = fmaf(xv.w, rw[(h + 3) * NEXP + e1], acc1);
    }
    // softmax over 64 logits (wave reduction)
    float m = fmaxf(acc0, acc1);
    for (int off = 16; off; off >>= 1) m = fmaxf(m, __shfl_xor(m, off, 32));
    float ex0 = __expf(acc0 - m), ex1 = __expf(acc1 - m);
    float ssum = ex0 + ex1;
    for (int off = 16; off; off >>= 1) ssum += __shfl_xor(ssum, off, 32);
    float p0 = ex0 / ssum, p1 = ex1 / ssum;

    // iterative top-4 (argmax with low-index tie-break), slot assignment
    int slot = 0;
#pragma unroll
    for (int k = 0; k < TOPK; ++k) {
        float bv = (p0 >= p1) ? p0 : p1;
        int   bx = (p0 >= p1) ? e0 : e1;
        for (int off = 16; off; off >>= 1) {
            float ov = __shfl_xor(bv, off, 32);
            int   oi = __shfl_xor(bx, off, 32);
            if (ov > bv || (ov == bv && oi < bx)) { bv = ov; bx = oi; }
        }
        if (lane == k) {
            int pos = atomicAdd(&cnt[bx], 1);
            if (pos > CAPE - 1) pos = CAPE - 1;   // safety clamp (mean load 512)
            slot = bx * CAPE + pos;
            smap[t * TOPK + k]  = slot;
            sprob[t * TOPK + k] = bv;
        }
        if (e0 == bx) p0 = -1.f;
        if (e1 == bx) p1 = -1.f;
    }
    // scatter token vector into each assigned expert slot (bf16)
#pragma unroll
    for (int k = 0; k < TOPK; ++k) {
        int sk = __shfl(slot, k, 32);
        unsigned short* dst = buf + (size_t)sk * HID;
        for (int h = lane; h < HID; h += 32) dst[h] = f2bf(xt[h]);
    }
}

// ---------------- kernel 3: grouped fc1 GEMM + fused SiLU-GLU ----------------
// tile: 128(M) x 64(N gate + paired up at +512); 8 wave32, 32x32 each
// double-buffered LDS; A staged by the Tensor Data Mover (one D# per K-slab,
// hardware pad recreates the 80B LDS row stride); B overlapped with WMMAs.
__global__ __launch_bounds__(256) void fc1_kernel(
    const unsigned short* __restrict__ buf, const unsigned short* __restrict__ w1b,
    const int* __restrict__ cnt, unsigned short* __restrict__ gbuf) {
    const int e  = blockIdx.x;
    const int m0 = blockIdx.y * 128;
    const int nt = blockIdx.z;                 // gate column block of 64
    if (m0 >= cnt[e]) return;                  // uniform early-exit (EXEC stays full)

    __shared__ __align__(16) unsigned short As[2][128 * AS];
    __shared__ __align__(16) unsigned short Bg[2][64 * BS];
    __shared__ __align__(16) unsigned short Bu[2][64 * BS];

    const int tid  = threadIdx.x;
    const int lane = tid & 31;
    const int w    = tid >> 5;
    const int wm   = w >> 1;                   // 0..3 (M)
    const int wn   = w & 1;                    // 0..1 (N)

    f32x8 accG[2][2] = {};
    f32x8 accU[2][2] = {};

    const unsigned short* Aglob = buf + ((size_t)e * CAPE + m0) * HID;
    const size_t baseB = (size_t)e * HID * N1;
    const int ng = nt * 64;                    // gate cols [ng, ng+64); up at +IMID

    // per-thread B staging coords
    const int whB = tid >> 7;                  // 0=gate, 1=up
    const int t2  = tid & 127;
    const int rB  = t2 >> 2, qB = t2 & 3;
    const int nbB = (whB ? (IMID + ng) : ng) + qB * 16;

    // ---- prologue: stage slab 0 ----
    if (w == 0)
        tdm_load_2d((unsigned)(size_t)&As[0][0], Aglob, HID, HID, 128, 32, 128);
    {
        const uint4* gp = (const uint4*)(w1b + baseB + (size_t)rB * N1 + nbB);
        uint4 b0 = gp[0], b1 = gp[1];
        unsigned short tmp[16];
        *(uint4*)&tmp[0] = b0; *(uint4*)&tmp[8] = b1;
        unsigned short* Bsel = whB ? Bu[0] : Bg[0];
#pragma unroll
        for (int j = 0; j < 16; ++j) Bsel[(qB * 16 + j) * BS + rB] = tmp[j];
    }
    if (w == 0) wait_tensor();
    __syncthreads();

    int cur = 0;
    for (int kk = 0; kk < HID; kk += 32, cur ^= 1) {
        const int  nxt  = cur ^ 1;
        const bool more = (kk + 32) < HID;
        uint4 b0 = {}, b1 = {};
        if (more) {
            if (w == 0)                                                // TDM next A slab
                tdm_load_2d((unsigned)(size_t)&As[nxt][0], Aglob + kk + 32,
                            HID, HID, 128, 32, 128);
            const uint4* gp = (const uint4*)(w1b + baseB + (size_t)(kk + 32 + rB) * N1 + nbB);
            b0 = gp[0]; b1 = gp[1];                                    // in flight over WMMAs
        }

        // ---- compute on current slab ----
        bf16x16 a0 = load_fragA(As[cur], wm * 32 + 0  + (lane & 15), lane);
        bf16x16 a1 = load_fragA(As[cur], wm * 32 + 16 + (lane & 15), lane);
        const int c0 = wn * 32 + (lane & 15);
        bf16x16 bg0 = load_fragB(Bg[cur], c0,      lane);
        bf16x16 bg1 = load_fragB(Bg[cur], c0 + 16, lane);
        bf16x16 bu0 = load_fragB(Bu[cur], c0,      lane);
        bf16x16 bu1 = load_fragB(Bu[cur], c0 + 16, lane);
        accG[0][0] = wmma_bf16(a0, bg0, accG[0][0]);
        accG[1][0] = wmma_bf16(a1, bg0, accG[1][0]);
        accG[0][1] = wmma_bf16(a0, bg1, accG[0][1]);
        accG[1][1] = wmma_bf16(a1, bg1, accG[1][1]);
        accU[0][0] = wmma_bf16(a0, bu0, accU[0][0]);
        accU[1][0] = wmma_bf16(a1, bu0, accU[1][0]);
        accU[0][1] = wmma_bf16(a0, bu1, accU[0][1]);
        accU[1][1] = wmma_bf16(a1, bu1, accU[1][1]);

        if (more) {                                                    // transpose-store B
            unsigned short tmp[16];
            *(uint4*)&tmp[0] = b0; *(uint4*)&tmp[8] = b1;
            unsigned short* Bsel = whB ? Bu[nxt] : Bg[nxt];
#pragma unroll
            for (int j = 0; j < 16; ++j) Bsel[(qB * 16 + j) * BS + rB] = tmp[j];
        }
        if (w == 0) wait_tensor();
        __syncthreads();
    }

    // combine: g = silu(gate) * up, write bf16 [E][CAP][I]
    const int colb = ng + wn * 32 + (lane & 15);
    const int rowb = m0 + wm * 32 + ((lane < 16) ? 0 : 8);
#pragma unroll
    for (int mi = 0; mi < 2; ++mi)
#pragma unroll
        for (int ni = 0; ni < 2; ++ni)
#pragma unroll
            for (int j = 0; j < 8; ++j) {
                float gv = accG[mi][ni][j];
                float uv = accU[mi][ni][j];
                float sv = gv * __builtin_amdgcn_rcpf(1.0f + __expf(-gv)) * uv;
                int row = rowb + mi * 16 + j;
                int col = colb + ni * 16;
                gbuf[((size_t)e * CAPE + row) * IMID + col] = f2bf(sv);
            }
}

// ---------------- kernel 4: grouped fc2 GEMM (async-LDS staging path) ----------------
__global__ __launch_bounds__(256) void fc2_kernel(
    const unsigned short* __restrict__ gbuf, const unsigned short* __restrict__ w2b,
    const int* __restrict__ cnt, unsigned short* __restrict__ ybuf) {
    const int e  = blockIdx.x;
    const int m0 = blockIdx.y * 128;
    const int n0 = blockIdx.z * 64;
    if (m0 >= cnt[e]) return;

    __shared__ __align__(16) unsigned short As[2][128 * AS];
    __shared__ __align__(16) unsigned short Bs[2][64 * BS];

    const int tid  = threadIdx.x;
    const int lane = tid & 31;
    const int w    = tid >> 5;
    const int wm   = w >> 1;
    const int wn   = w & 1;

    f32x8 acc[2][2] = {};
    const unsigned short* Aglob = gbuf + ((size_t)e * CAPE + m0) * IMID;
    const size_t baseB = (size_t)e * IMID * HID;

    const int rB = tid >> 3, qB = tid & 7;     // 8 shorts / thread

    // ---- prologue ----
    stageA_async(As[0], Aglob, IMID, tid, 0);
    {
        uint4 b0 = *(const uint4*)(w2b + baseB + (size_t)rB * HID + n0 + qB * 8);
        unsigned short tmp[8];
        *(uint4*)&tmp[0] = b0;
#pragma unroll
        for (int j = 0; j < 8; ++j) Bs[0][(qB * 8 + j) * BS + rB] = tmp[j];
    }
    wait_async();
    __syncthreads();

    int cur = 0;
    for (int kk = 0; kk < IMID; kk += 32, cur ^= 1) {
        const int  nxt  = cur ^ 1;
        const bool more = (kk + 32) < IMID;
        uint4 b0 = {};
        if (more) {
            stageA_async(As[nxt], Aglob, IMID, tid, kk + 32);
            b0 = *(const uint4*)(w2b + baseB + (size_t)(kk + 32 + rB) * HID + n0 + qB * 8);
        }

        bf16x16 a0 = load_fragA(As[cur], wm * 32 + 0  + (lane & 15), lane);
        bf16x16 a1 = load_fragA(As[cur], wm * 32 + 16 + (lane & 15), lane);
        const int c0 = wn * 32 + (lane & 15);
        bf16x16 f0 = load_fragB(Bs[cur], c0,      lane);
        bf16x16 f1 = load_fragB(Bs[cur], c0 + 16, lane);
        acc[0][0] = wmma_bf16(a0, f0, acc[0][0]);
        acc[1][0] = wmma_bf16(a1, f0, acc[1][0]);
        acc[0][1] = wmma_bf16(a0, f1, acc[0][1]);
        acc[1][1] = wmma_bf16(a1, f1, acc[1][1]);

        if (more) {
            unsigned short tmp[8];
            *(uint4*)&tmp[0] = b0;
#pragma unroll
            for (int j = 0; j < 8; ++j) Bs[nxt][(qB * 8 + j) * BS + rB] = tmp[j];
        }
        wait_async();
        __syncthreads();
    }

    const int colb = n0 + wn * 32 + (lane & 15);
    const int rowb = m0 + wm * 32 + ((lane < 16) ? 0 : 8);
#pragma unroll
    for (int mi = 0; mi < 2; ++mi)
#pragma unroll
        for (int ni = 0; ni < 2; ++ni)
#pragma unroll
            for (int j = 0; j < 8; ++j) {
                int row = rowb + mi * 16 + j;
                int col = colb + ni * 16;
                ybuf[((size_t)e * CAPE + row) * HID + col] = f2bf(acc[mi][ni][j]);
            }
}

// ---------------- kernel 5: gather un-permute (prob-weighted sum) ----------------
__global__ __launch_bounds__(256) void unpermute_kernel(
    const unsigned short* __restrict__ ybuf, const int* __restrict__ smap,
    const float* __restrict__ sprob, float* __restrict__ out) {
    const int t   = blockIdx.x;
    const int tid = threadIdx.x;
    const int s0 = smap[t * TOPK + 0], s1 = smap[t * TOPK + 1];
    const int s2 = smap[t * TOPK + 2], s3 = smap[t * TOPK + 3];
    const float p0 = sprob[t * TOPK + 0], p1 = sprob[t * TOPK + 1];
    const float p2 = sprob[t * TOPK + 2], p3 = sprob[t * TOPK + 3];
#pragma unroll
    for (int j = 0; j < 4; ++j) {
        int h = tid + j * 256;
        float acc = p0 * bf2f(ybuf[(size_t)s0 * HID + h]);
        acc = fmaf(p1, bf2f(ybuf[(size_t)s1 * HID + h]), acc);
        acc = fmaf(p2, bf2f(ybuf[(size_t)s2 * HID + h]), acc);
        acc = fmaf(p3, bf2f(ybuf[(size_t)s3 * HID + h]), acc);
        out[(size_t)t * HID + h] = acc;
    }
}

// ---------------- launch ----------------
extern "C" void kernel_launch(void* const* d_in, const int* in_sizes, int n_in,
                              void* d_out, int out_size, void* d_ws, size_t ws_size,
                              hipStream_t stream) {
    const float* x  = (const float*)d_in[0];   // [S,B,H]
    const float* rw = (const float*)d_in[1];   // [H,E]
    const float* w1 = (const float*)d_in[2];   // [E,H,2I]
    const float* w2 = (const float*)d_in[3];   // [E,I,H]
    float* out = (float*)d_out;

    char* ws = (char*)d_ws;
    int*            cnt   = (int*)(ws + OFF_CNT);
    int*            smap  = (int*)(ws + OFF_SMAP);
    float*          sprob = (float*)(ws + OFF_SPROB);
    unsigned short* w1b   = (unsigned short*)(ws + OFF_W1B);
    unsigned short* w2b   = (unsigned short*)(ws + OFF_W2B);
    unsigned short* buf   = (unsigned short*)(ws + OFF_BUF);
    unsigned short* gbuf  = (unsigned short*)(ws + OFF_G);
    unsigned short* ybuf  = (unsigned short*)(ws + OFF_Y);

    convert_init_kernel<<<1024, 256, 0, stream>>>(w1, w2, w1b, w2b, cnt);
    router_kernel<<<T_ / 8, 256, 0, stream>>>(x, rw, cnt, smap, sprob, buf);
    fc1_kernel<<<dim3(NEXP, CAPE / 128, IMID / 64), 256, 0, stream>>>(buf, w1b, cnt, gbuf);
    fc2_kernel<<<dim3(NEXP, CAPE / 128, HID / 64), 256, 0, stream>>>(gbuf, w2b, cnt, ybuf);
    unpermute_kernel<<<T_, 256, 0, stream>>>(ybuf, smap, sprob, out);
}
